// ShotNCHead_24180665876742
// MI455X (gfx1250) — matrix operand order repaired and checked by reading, not compiled, not run on hardware
//
#include <hip/hip_runtime.h>
#include <hip/hip_bf16.h>

// ---------------------------------------------------------------------------
// GCN 2-layer inference + cosine head for MI455X (gfx1250, wave32, WMMA).
// GEMMs: bf16 V_WMMA_F32_16X16X32_BF16 (f32 accum). Scatter: wave-per-edge
// float4 gather + global f32 atomics (L2-resident working set).
// ---------------------------------------------------------------------------

typedef __attribute__((ext_vector_type(16))) __bf16 v16bf;
typedef __attribute__((ext_vector_type(8)))  float  v8f;

#define D_IN  256
#define D_H   128
#define D_C   128
#define NCLS  40

union BFrag { v16bf v; uint4 u[2]; };

// ---- init: zero accumulators, deg = 1.0 (self-loop contribution) ----------
__global__ void k_init(float* __restrict__ agg1, float* __restrict__ agg2,
                       float* __restrict__ deg, int n) {
  long long i = (long long)blockIdx.x * blockDim.x + threadIdx.x;
  long long tot = (long long)n * D_H;
  if (i < tot) { agg1[i] = 0.0f; agg2[i] = 0.0f; }
  if (i < n) deg[i] = 1.0f;
}

// ---- f32 -> bf16 conversion ------------------------------------------------
__global__ void k_cvt_bf16(const float* __restrict__ src, __bf16* __restrict__ dst,
                           long long n) {
  long long i = (long long)blockIdx.x * blockDim.x + threadIdx.x;
  if (i < n) dst[i] = (__bf16)src[i];
}

// ---- W[k][n] -> Wt[n][k] bf16 (column-major so B frags load contiguously) --
__global__ void k_cvt_wT(const float* __restrict__ w, __bf16* __restrict__ wt,
                         int K, int N2) {
  int i = blockIdx.x * blockDim.x + threadIdx.x;
  if (i < K * N2) {
    int n = i / K, k = i % K;
    wt[i] = (__bf16)w[(long long)k * N2 + n];
  }
}

// ---- degree: atomic count per destination node -----------------------------
__global__ void k_degree(const long long* __restrict__ col, float* __restrict__ deg,
                         int E) {
  int e = blockIdx.x * blockDim.x + threadIdx.x;
  if (e < E) atomicAdd(&deg[col[e]], 1.0f);
}

// ---- dinv = rsqrt(deg) in place --------------------------------------------
__global__ void k_dinv(float* __restrict__ deg, int n) {
  int i = blockIdx.x * blockDim.x + threadIdx.x;
  if (i < n) { float d = deg[i]; deg[i] = d > 0.0f ? rsqrtf(d) : 0.0f; }
}

// ---- bf16 WMMA GEMM: C[M x N2] = A[M x K] * Bt[N2 x K]^T -------------------
// One wave per 16x16 output tile; 8 waves/block cover the full 128-wide N.
template <int K>
__global__ __launch_bounds__(256)
void k_gemm_bf16(const __bf16* __restrict__ A, const __bf16* __restrict__ Bt,
                 float* __restrict__ C, int M, int N2) {
  const int lane = threadIdx.x & 31;
  const int wave = threadIdx.x >> 5;
  const int m0 = blockIdx.x * 16;
  const int n0 = wave * 16;
  const int half = lane >> 4;   // 0: lanes 0-15, 1: lanes 16-31
  const int l16  = lane & 15;

  // A layout (matches B): lane<16 holds K 0..15; lane>=16 holds K 16..31
  int arow = m0 + l16; if (arow >= M) arow = M - 1;           // keep EXEC all-1s
  const __bf16* Ap = A  + (long long)arow * K + half * 16;
  // B layout (matches A): lane<16 holds K 0..15; lane>=16 holds K 16..31
  const __bf16* Bp = Bt + (long long)(n0 + l16) * K + half * 16;

  v8f acc = {};
#pragma unroll
  for (int k0 = 0; k0 < K; k0 += 32) {
    BFrag a, b;
    a.u[0] = *(const uint4*)(Ap + k0);        // 8 contiguous bf16
    a.u[1] = *(const uint4*)(Ap + k0 + 8);
    b.u[0] = *(const uint4*)(Bp + k0);
    b.u[1] = *(const uint4*)(Bp + k0 + 8);
    acc = __builtin_amdgcn_wmma_f32_16x16x32_bf16(
        false, a.v, false, b.v, (short)0, acc, false, false);
  }

  // C/D layout: VGPR r -> row m0 + r (+8 for upper lane half); col = n0 + l16
  float* Cp = C + (long long)(m0 + half * 8) * N2 + n0 + l16;
#pragma unroll
  for (int r = 0; r < 8; ++r) {
    int m = m0 + half * 8 + r;
    if (m < M) Cp[(long long)r * N2] = acc[r];
  }
}

// ---- edge scatter: AGG[col] += H[row] * dinv[row]*dinv[col] ----------------
// One wave32 per edge: lane does one float4 gather + 4 f32 atomics (L2-resident).
__global__ __launch_bounds__(256)
void k_scatter(const long long* __restrict__ row, const long long* __restrict__ col,
               const float* __restrict__ dinv, const float* __restrict__ H,
               float* __restrict__ AGG, int E) {
  int wid  = (int)(((long long)blockIdx.x * blockDim.x + threadIdx.x) >> 5);
  int lane = threadIdx.x & 31;
  if (wid >= E) return;
  long long r = row[wid];            // wave-uniform -> scalar load
  long long c = col[wid];
  if (wid + 2048 < E) {              // prefetch a future gather row into L2
    long long rp = row[wid + 2048];
    __builtin_prefetch(H + rp * D_H, 0, 0);   // global_prefetch_b8
  }
  float nrm = dinv[r] * dinv[c];
  float4 v = ((const float4*)(H + r * D_H))[lane];
  float* dst = AGG + c * D_H + lane * 4;
  atomicAdd(dst + 0, v.x * nrm);
  atomicAdd(dst + 1, v.y * nrm);
  atomicAdd(dst + 2, v.z * nrm);
  atomicAdd(dst + 3, v.w * nrm);
}

// ---- epilogue 1: add self-loop msg + bias, ReLU, quantize to bf16 ----------
__global__ void k_post1(const float* __restrict__ agg1, const float* __restrict__ h1,
                        const float* __restrict__ dinv, const float* __restrict__ b1,
                        __bf16* __restrict__ h1r, int n) {
  long long i = (long long)blockIdx.x * blockDim.x + threadIdx.x;
  if (i < (long long)n * D_H) {
    int node = (int)(i >> 7);
    int f = (int)(i & (D_H - 1));
    float di = dinv[node];
    float v = agg1[i] + h1[i] * di * di + b1[f];
    h1r[i] = (__bf16)fmaxf(v, 0.0f);
  }
}

// ---- normalize class embeddings: clsn[c] = cls[c]/max(||cls[c]||,1e-8) -----
__global__ void k_clsnorm(const float* __restrict__ cls, float* __restrict__ clsn) {
  int wave = threadIdx.x >> 5, lane = threadIdx.x & 31;
  int c = blockIdx.x * 8 + wave;
  if (c >= NCLS) return;
  float4 v = ((const float4*)(cls + c * D_C))[lane];
  float ss = v.x * v.x + v.y * v.y + v.z * v.z + v.w * v.w;
#pragma unroll
  for (int m = 16; m >= 1; m >>= 1) ss += __shfl_xor(ss, m, 32);
  float inv = 1.0f / fmaxf(sqrtf(ss), 1e-8f);
  float4 o = { v.x * inv, v.y * inv, v.z * inv, v.w * inv };
  ((float4*)(clsn + c * D_C))[lane] = o;
}

// ---- epilogue 2 + head: h = agg2 + self-loop + b2; out = (h/||h||)·clsn^T --
__global__ __launch_bounds__(256)
void k_head(const float* __restrict__ agg2, const float* __restrict__ h2,
            const float* __restrict__ dinv, const float* __restrict__ b2,
            const float* __restrict__ clsn, float* __restrict__ out, int n) {
  __shared__ float scls[NCLS * D_C];   // 20 KB of the 320 KB WGP LDS
  for (int i = threadIdx.x; i < NCLS * D_C; i += blockDim.x) scls[i] = clsn[i];
  __syncthreads();

  int wave = threadIdx.x >> 5, lane = threadIdx.x & 31;
  int node = blockIdx.x * 8 + wave;
  if (node >= n) return;

  float di = dinv[node];
  float4 a4 = ((const float4*)(agg2 + (long long)node * D_C))[lane];
  float4 h4 = ((const float4*)(h2   + (long long)node * D_C))[lane];
  float4 b4 = ((const float4*)b2)[lane];
  float4 h;
  h.x = a4.x + h4.x * di * di + b4.x;
  h.y = a4.y + h4.y * di * di + b4.y;
  h.z = a4.z + h4.z * di * di + b4.z;
  h.w = a4.w + h4.w * di * di + b4.w;

  float ss = h.x * h.x + h.y * h.y + h.z * h.z + h.w * h.w;
#pragma unroll
  for (int m = 16; m >= 1; m >>= 1) ss += __shfl_xor(ss, m, 32);
  float inv = 1.0f / fmaxf(sqrtf(ss), 1e-8f);

  for (int c = 0; c < NCLS; ++c) {
    float4 c4 = ((const float4*)(scls + c * D_C))[lane];
    float d = h.x * c4.x + h.y * c4.y + h.z * c4.z + h.w * c4.w;
#pragma unroll
    for (int m = 16; m >= 1; m >>= 1) d += __shfl_xor(d, m, 32);
    if (lane == 0) out[(long long)node * NCLS + c] = d * inv;
  }
}

// ---------------------------------------------------------------------------
extern "C" void kernel_launch(void* const* d_in, const int* in_sizes, int n_in,
                              void* d_out, int out_size, void* d_ws, size_t ws_size,
                              hipStream_t stream) {
  const float*     x   = (const float*)d_in[0];
  const long long* ei  = (const long long*)d_in[1];   // int64 edge_index [2,E]
  const float*     W1  = (const float*)d_in[2];
  const float*     b1  = (const float*)d_in[3];
  const float*     W2  = (const float*)d_in[4];
  const float*     b2  = (const float*)d_in[5];
  const float*     cls = (const float*)d_in[6];
  float*           out = (float*)d_out;

  const int N = in_sizes[0] / D_IN;
  const int E = in_sizes[1] / 2;
  const long long* row = ei;
  const long long* col = ei + E;

  // ---- carve workspace (~283 MB) ----
  char* p = (char*)d_ws;
  auto carve = [&](size_t bytes) -> char* {
    char* q = p; p += (bytes + 255) & ~(size_t)255; return q;
  };
  __bf16* xbf  = (__bf16*)carve((size_t)N * D_IN * 2);
  __bf16* w1t  = (__bf16*)carve((size_t)D_IN * D_H * 2);
  __bf16* w2t  = (__bf16*)carve((size_t)D_H * D_C * 2);
  float*  h1   = (float*) carve((size_t)N * D_H * 4);
  float*  agg1 = (float*) carve((size_t)N * D_H * 4);
  __bf16* h1r  = (__bf16*)carve((size_t)N * D_H * 2);
  float*  h2   = (float*) carve((size_t)N * D_C * 4);
  float*  agg2 = (float*) carve((size_t)N * D_C * 4);
  float*  deg  = (float*) carve((size_t)N * 4);       // becomes dinv in place
  float*  clsn = (float*) carve((size_t)NCLS * D_C * 4);

  const int T = 256;
  const long long nd = (long long)N * D_H;

  k_init    <<<(unsigned)((nd + T - 1) / T), T, 0, stream>>>(agg1, agg2, deg, N);
  k_cvt_bf16<<<(unsigned)(((long long)N * D_IN + T - 1) / T), T, 0, stream>>>(
      x, xbf, (long long)N * D_IN);
  k_cvt_wT  <<<(D_IN * D_H + T - 1) / T, T, 0, stream>>>(W1, w1t, D_IN, D_H);
  k_cvt_wT  <<<(D_H * D_C + T - 1) / T, T, 0, stream>>>(W2, w2t, D_H, D_C);
  k_clsnorm <<<(NCLS + 7) / 8, T, 0, stream>>>(cls, clsn);
  k_degree  <<<(E + T - 1) / T, T, 0, stream>>>(col, deg, E);
  k_dinv    <<<(N + T - 1) / T, T, 0, stream>>>(deg, N);

  // layer 1: h1 = x @ W1 (bf16 WMMA), scatter, bias+ReLU -> bf16
  k_gemm_bf16<D_IN><<<(N + 15) / 16, T, 0, stream>>>(xbf, w1t, h1, N, D_H);
  k_scatter <<<(E + 7) / 8, T, 0, stream>>>(row, col, deg, h1, agg1, E);
  k_post1   <<<(unsigned)((nd + T - 1) / T), T, 0, stream>>>(agg1, h1, deg, b1, h1r, N);

  // layer 2: h2 = h1r @ W2 (bf16 WMMA), scatter, bias + cosine head
  k_gemm_bf16<D_H><<<(N + 15) / 16, T, 0, stream>>>(h1r, w2t, h2, N, D_C);
  k_scatter <<<(E + 7) / 8, T, 0, stream>>>(row, col, deg, h2, agg2, E);
  k_head    <<<(N + 7) / 8, T, 0, stream>>>(agg2, h2, deg, b2, clsn, out, N);
}
